// MultiHeadAttention_24816321036508
// MI455X (gfx1250) — matrix operand order repaired and checked by reading, not compiled, not run on hardware
//
#include <hip/hip_runtime.h>
#include <hip/hip_bf16.h>

// ---------------------------------------------------------------------------
// Problem constants (from the reference)
// ---------------------------------------------------------------------------
constexpr int BATCH   = 2;
constexpr int SEQ     = 2048;
constexpr int HID     = 4096;
constexpr int NHEADS  = 32;
constexpr int NKV     = 8;
constexpr int HD      = 128;
constexpr int TOKENS  = BATCH * SEQ;                 // 4096
constexpr int QKVN    = NHEADS * HD + 2 * NKV * HD;  // 6144
constexpr float ATTN_MULT = 0.08838834764831845f;
constexpr float MAX_ATTN  = 30.0f;
constexpr float ROPE_BASE = 10000.0f;

// ---------------------------------------------------------------------------
// WMMA types / helpers (CDNA5 gfx1250, wave32)
// ---------------------------------------------------------------------------
typedef __attribute__((ext_vector_type(16))) __bf16 v16bf;
typedef __attribute__((ext_vector_type(8)))  float  v8f;
typedef int v4i_t __attribute__((vector_size(16)));

static __device__ __forceinline__ unsigned short f2bf(float f) {
  union { float f; unsigned u; } v; v.f = f;
  unsigned r = v.u + 0x7FFFu + ((v.u >> 16) & 1u);   // round-to-nearest-even
  return (unsigned short)(r >> 16);
}
static __device__ __forceinline__ float bf2f(unsigned short h) {
  union { unsigned u; float f; } v; v.u = ((unsigned)h) << 16; return v.f;
}

// Build a 16-element bf16 fragment from two 16-byte chunks.
static __device__ __forceinline__ v16bf frag2(const unsigned short* p0,
                                              const unsigned short* p1) {
  union { uint4 u[2]; v16bf v; } x;
  x.u[0] = *(const uint4*)p0;
  x.u[1] = *(const uint4*)p1;
  return x.v;
}

// A-matrix fragment (16x32, row-major source, row = lane&15):
//   lane group g holds K = {g*8 .. g*8+7} and {16+g*8 .. 16+g*8+7}
static __device__ __forceinline__ v16bf load_a_frag(const unsigned short* rowPtr,
                                                    int g) {
  const unsigned short* p = rowPtr + g * 8;
  return frag2(p, p + 16);
}
// B-matrix fragment (32x16, K-contiguous source per column, col = lane&15):
//   lane group g holds K = {g*8 .. g*8+7} and {16+g*8 .. 16+g*8+7}
static __device__ __forceinline__ v16bf load_b_frag(const unsigned short* colPtr,
                                                    int g) {
  const unsigned short* p = colPtr + g * 8;
  return frag2(p, p + 16);
}

static __device__ __forceinline__ v8f wmma_bf16(v16bf a, v16bf b, v8f c) {
  return __builtin_amdgcn_wmma_f32_16x16x32_bf16(
      /*neg_a=*/false, a, /*neg_b=*/false, b,
      /*c_mod=*/(short)0, c, /*reuse_a=*/false, /*reuse_b=*/false);
}

static __device__ __forceinline__ v8f vzero8() {
  v8f z;
#pragma unroll
  for (int i = 0; i < 8; ++i) z[i] = 0.0f;
  return z;
}

// ---------------------------------------------------------------------------
// Async global->LDS copy (CDNA5 async-DMA path). 16 bytes per lane.
// Falls back to a synchronous load+ds_store if the builtin is unavailable.
// ---------------------------------------------------------------------------
#if __has_builtin(__builtin_amdgcn_global_load_async_to_lds_b128)
#define HAVE_ASYNC_LDS 1
#else
#define HAVE_ASYNC_LDS 0
#endif

static __device__ __forceinline__ void async_copy16(const unsigned short* g,
                                                    unsigned short* l) {
#if HAVE_ASYNC_LDS
  // Param types (from clang diagnostics): non-const v4i32* in device (AS1)
  // and shared (AS3) address spaces. Integer round-trips avoid pointer
  // address-space cast restrictions; LDS generic addr low 32 bits = offset.
  __builtin_amdgcn_global_load_async_to_lds_b128(
      (__attribute__((address_space(1))) v4i_t*)(unsigned long long)g,
      (__attribute__((address_space(3))) v4i_t*)(unsigned int)(unsigned long long)l,
      0, 0);
#else
  *(uint4*)l = *(const uint4*)g;
#endif
}
static __device__ __forceinline__ void wait_async_le8() {
#if HAVE_ASYNC_LDS
#if __has_builtin(__builtin_amdgcn_s_wait_asynccnt)
  __builtin_amdgcn_s_wait_asynccnt(8);
#else
  asm volatile("s_wait_asynccnt 8" ::: "memory");
#endif
#endif
}
static __device__ __forceinline__ void wait_async_0() {
#if HAVE_ASYNC_LDS
#if __has_builtin(__builtin_amdgcn_s_wait_asynccnt)
  __builtin_amdgcn_s_wait_asynccnt(0);
#else
  asm volatile("s_wait_asynccnt 0" ::: "memory");
#endif
#endif
}

// ---------------------------------------------------------------------------
// Elementwise conversion kernels
// ---------------------------------------------------------------------------
__global__ void k_f32_to_bf16(const float* __restrict__ src,
                              unsigned short* __restrict__ dst, int n) {
  int i = blockIdx.x * blockDim.x + threadIdx.x;
  if (i < n) dst[i] = f2bf(src[i]);
}

// W [K][N] row-major f32  ->  Wt [N][K] row-major bf16
__global__ void k_transpose_bf16(const float* __restrict__ W,
                                 unsigned short* __restrict__ Wt,
                                 int K, int N) {
  int i = blockIdx.x * blockDim.x + threadIdx.x;
  if (i >= K * N) return;
  int k = i / N, n = i - k * N;
  Wt[(size_t)n * K + k] = f2bf(W[(size_t)k * N + n]);
}

// ---------------------------------------------------------------------------
// Tiled bf16 WMMA GEMM: C[M,N] = A[M,K] * Bt[N,K]^T
// Block: 256 threads (8 waves), tile 128(M) x 128(N), K-step 64.
// Double-buffered LDS staging filled by async global->LDS copies.
// Each wave computes a 32x64 sub-tile (2x4 WMMA accumulators).
// ---------------------------------------------------------------------------
template <bool BF16OUT>
__global__ __launch_bounds__(256) void k_gemm_bf16(
    const unsigned short* __restrict__ A, const unsigned short* __restrict__ Bt,
    void* __restrict__ Cv, int M, int N, int K) {
  constexpr int KSTEP   = 64;
  constexpr int LSTRIDE = 72;  // 64 data + 8 pad shorts = 144B (16B aligned, bank-spread)
  __shared__ alignas(16) unsigned short aL[2][128 * LSTRIDE];
  __shared__ alignas(16) unsigned short bL[2][128 * LSTRIDE];

  const int tid  = threadIdx.x;
  const int lane = tid & 31;
  const int wave = tid >> 5;
  const int g = lane >> 4, idx = lane & 15;
  const int rowbase = (wave >> 1) * 32;  // 0,32,64,96
  const int colbase = (wave & 1) * 64;   // 0,64
  const int gRow = blockIdx.y * 128;
  const int gCol = blockIdx.x * 128;

  // One stage = 128x64 bf16 tile of A and of B = 2 x 1024 16B-chunks.
  // 256 threads x 4 iterations x (1 A-chunk + 1 B-chunk) => 8 async ops/wave.
  auto issue_stage = [&](int stage, int buf) {
    const int k0 = stage * KSTEP;
#pragma unroll
    for (int i = 0; i < 4; ++i) {
      int c = tid + i * 256;      // 0..1023
      int row = c >> 3;           // 0..127
      int part = c & 7;           // 8 x 16B chunks per 64-element row
      async_copy16(A + (size_t)(gRow + row) * K + k0 + part * 8,
                   &aL[buf][row * LSTRIDE + part * 8]);
      async_copy16(Bt + (size_t)(gCol + row) * K + k0 + part * 8,
                   &bL[buf][row * LSTRIDE + part * 8]);
    }
  };

  v8f acc[2][4];
#pragma unroll
  for (int mt = 0; mt < 2; ++mt)
#pragma unroll
    for (int nt = 0; nt < 4; ++nt) acc[mt][nt] = vzero8();

  const int nk = K / KSTEP;
  issue_stage(0, 0);
  for (int it = 0; it < nk; ++it) {
    const int buf = it & 1;
    if (it + 1 < nk) {
      issue_stage(it + 1, buf ^ 1);  // overlap next fill with this compute
      wait_async_le8();              // retire stage `it` (in-order completion)
    } else {
      wait_async_0();
    }
    __syncthreads();
#pragma unroll
    for (int kc = 0; kc < 2; ++kc) {
      v16bf af[2];
#pragma unroll
      for (int mt = 0; mt < 2; ++mt) {
        const unsigned short* p =
            &aL[buf][(rowbase + mt * 16 + idx) * LSTRIDE + kc * 32 + g * 8];
        af[mt] = frag2(p, p + 16);
      }
      v16bf bfr[4];
#pragma unroll
      for (int nt = 0; nt < 4; ++nt) {
        const unsigned short* p =
            &bL[buf][(colbase + nt * 16 + idx) * LSTRIDE + kc * 32 + g * 8];
        bfr[nt] = frag2(p, p + 16);
      }
#pragma unroll
      for (int mt = 0; mt < 2; ++mt)
#pragma unroll
        for (int nt = 0; nt < 4; ++nt)
          acc[mt][nt] = wmma_bf16(af[mt], bfr[nt], acc[mt][nt]);
    }
    __syncthreads();  // all waves done reading before next overwrite
  }

#pragma unroll
  for (int mt = 0; mt < 2; ++mt)
#pragma unroll
    for (int nt = 0; nt < 4; ++nt)
#pragma unroll
      for (int r = 0; r < 8; ++r) {
        int row = gRow + rowbase + mt * 16 + r + 8 * g;
        int col = gCol + colbase + nt * 16 + idx;
        size_t o = (size_t)row * N + col;
        if (BF16OUT) ((unsigned short*)Cv)[o] = f2bf(acc[mt][nt][r]);
        else         ((float*)Cv)[o]          = acc[mt][nt][r];
      }
}

// ---------------------------------------------------------------------------
// RoPE + scatter: qkv [TOKENS][QKVN] bf16 -> head-major buffers
// ---------------------------------------------------------------------------
__global__ void k_rope(const unsigned short* __restrict__ qkv,
                       unsigned short* __restrict__ out,
                       int nheads, int colOff, int total) {
  int t = blockIdx.x * blockDim.x + threadIdx.x;
  if (t >= total) return;
  int i   = t & 63;
  int hh  = (t >> 6) % nheads;
  int tok = t / (64 * nheads);
  int s   = tok % SEQ;
  int b   = tok / SEQ;
  int col = colOff + hh * HD + i;
  float q0 = bf2f(qkv[(size_t)tok * QKVN + col]);
  float q1 = bf2f(qkv[(size_t)tok * QKVN + col + 64]);
  float invf = __powf(ROPE_BASE, -(float)(2 * i) / (float)HD);
  float ang = (float)s * invf;
  float sn, cs;
  __sincosf(ang, &sn, &cs);
  float r0 = q0 * cs - q1 * sn;
  float r1 = q1 * cs + q0 * sn;
  size_t base = ((size_t)(b * nheads + hh) * SEQ + s) * HD;
  out[base + i]      = f2bf(r0);
  out[base + i + 64] = f2bf(r1);
}

// V transpose: qkv V columns -> Vt [B][NKV][HD][SEQ]
__global__ void k_scatter_v(const unsigned short* __restrict__ qkv,
                            unsigned short* __restrict__ Vt, int total) {
  int t = blockIdx.x * blockDim.x + threadIdx.x;
  if (t >= total) return;
  int d   = t & (HD - 1);
  int kh  = (t >> 7) & (NKV - 1);
  int tok = t >> 10;
  int s   = tok % SEQ;
  int b   = tok / SEQ;
  Vt[((size_t)(b * NKV + kh) * HD + d) * SEQ + s] =
      qkv[(size_t)tok * QKVN + NHEADS * HD + NKV * HD + kh * HD + d];
}

// ---------------------------------------------------------------------------
// Flash attention: one wave per (b, h, 16-query tile).
// Q [B][NH][S][HD] bf16, K [B][NKV][S][HD] bf16, Vt [B][NKV][HD][S] bf16.
// Online softmax with tanh soft-cap and causal mask (+ -1e9, as reference).
// ---------------------------------------------------------------------------
__global__ __launch_bounds__(32) void k_attn(
    const unsigned short* __restrict__ Q, const unsigned short* __restrict__ Kc,
    const unsigned short* __restrict__ Vt, unsigned short* __restrict__ Aout) {
  const int lane = threadIdx.x & 31;
  const int g = lane >> 4, idx = lane & 15;
  const int bid = blockIdx.x;
  const int qt = bid & (SEQ / 16 - 1);        // 0..127
  const int h  = (bid >> 7) & (NHEADS - 1);   // 0..31
  const int b  = bid >> 12;                   // 0..1
  const int kvh = h >> 2;                     // GQA: 4 Q heads per KV head

  const unsigned short* Qb = Q  + (((size_t)(b * NHEADS + h) * SEQ) + qt * 16) * HD;
  const unsigned short* Kb = Kc + ((size_t)(b * NKV + kvh)) * SEQ * HD;
  const unsigned short* Vb = Vt + ((size_t)(b * NKV + kvh)) * HD * SEQ;

  // Resident Q fragments: 16 rows x 128 dims = 4 A-frags of 16x32
  v16bf qf[4];
#pragma unroll
  for (int kc = 0; kc < 4; ++kc)
    qf[kc] = load_a_frag(Qb + (size_t)idx * HD + kc * 32, g);

  v8f o[8];
#pragma unroll
  for (int t = 0; t < 8; ++t) o[t] = vzero8();
  float m[8], l[8];
#pragma unroll
  for (int r = 0; r < 8; ++r) { m[r] = -1e38f; l[r] = 0.0f; }

  __shared__ alignas(16) unsigned short pLds[16 * 32];

  const int qend = qt * 16 + 16;
  const int nkt = (qend + 31) / 32;
  for (int kt = 0; kt < nkt; ++kt) {
    const int kb = kt * 32;
    // ---- scores S = Q * K^T  (16 x 32, two 16-key column groups) ----
    v8f s0 = vzero8(), s1 = vzero8();
#pragma unroll
    for (int kc = 0; kc < 4; ++kc) {
      v16bf bk = load_b_frag(Kb + (size_t)(kb + idx) * HD + kc * 32, g);
      s0 = wmma_bf16(qf[kc], bk, s0);
    }
#pragma unroll
    for (int kc = 0; kc < 4; ++kc) {
      v16bf bk = load_b_frag(Kb + (size_t)(kb + 16 + idx) * HD + kc * 32, g);
      s1 = wmma_bf16(qf[kc], bk, s1);
    }
    // ---- soft-cap + causal mask + online softmax ----
    float rmx[8];
#pragma unroll
    for (int r = 0; r < 8; ++r) {
      int row = qt * 16 + r + 8 * g;
      float x0 = MAX_ATTN * tanhf(s0[r] * (ATTN_MULT / MAX_ATTN));
      float x1 = MAX_ATTN * tanhf(s1[r] * (ATTN_MULT / MAX_ATTN));
      if (kb + idx      > row) x0 += -1e9f;
      if (kb + 16 + idx > row) x1 += -1e9f;
      s0[r] = x0; s1[r] = x1;
      rmx[r] = fmaxf(x0, x1);
    }
#pragma unroll
    for (int off = 1; off < 16; off <<= 1)
#pragma unroll
      for (int r = 0; r < 8; ++r)
        rmx[r] = fmaxf(rmx[r], __shfl_xor(rmx[r], off, 32));

    float scale[8], lad[8];
#pragma unroll
    for (int r = 0; r < 8; ++r) {
      float mn = fmaxf(m[r], rmx[r]);
      scale[r] = __expf(m[r] - mn);
      float p0 = __expf(s0[r] - mn);
      float p1 = __expf(s1[r] - mn);
      s0[r] = p0; s1[r] = p1;
      m[r] = mn;
      lad[r] = p0 + p1;
    }
#pragma unroll
    for (int off = 1; off < 16; off <<= 1)
#pragma unroll
      for (int r = 0; r < 8; ++r) lad[r] += __shfl_xor(lad[r], off, 32);
#pragma unroll
    for (int r = 0; r < 8; ++r) l[r] = l[r] * scale[r] + lad[r];
#pragma unroll
    for (int t = 0; t < 8; ++t)
#pragma unroll
      for (int r = 0; r < 8; ++r) o[t][r] *= scale[r];

    // ---- P (C-layout) -> LDS -> A-frag, then O += P * V ----
    __syncthreads();
#pragma unroll
    for (int r = 0; r < 8; ++r) {
      pLds[(r + 8 * g) * 32 + idx]      = f2bf(s0[r]);
      pLds[(r + 8 * g) * 32 + 16 + idx] = f2bf(s1[r]);
    }
    __syncthreads();
    v16bf pf = load_a_frag(&pLds[idx * 32], g);
#pragma unroll
    for (int t = 0; t < 8; ++t) {
      v16bf vf = load_b_frag(Vb + (size_t)(t * 16 + idx) * SEQ + kb, g);
      o[t] = wmma_bf16(pf, vf, o[t]);
    }
  }

  // ---- epilogue: normalize and store [token][NH*HD] ----
  float inv[8];
#pragma unroll
  for (int r = 0; r < 8; ++r) inv[r] = 1.0f / l[r];
#pragma unroll
  for (int t = 0; t < 8; ++t)
#pragma unroll
    for (int r = 0; r < 8; ++r) {
      int row = qt * 16 + r + 8 * g;
      Aout[((size_t)b * SEQ + row) * (NHEADS * HD) + h * HD + t * 16 + idx] =
          f2bf(o[t][r] * inv[r]);
    }
}

// ---------------------------------------------------------------------------
// Host launch
// ---------------------------------------------------------------------------
extern "C" void kernel_launch(void* const* d_in, const int* in_sizes, int n_in,
                              void* d_out, int out_size, void* d_ws,
                              size_t ws_size, hipStream_t stream) {
  (void)in_sizes; (void)n_in; (void)out_size; (void)ws_size;
  const float* hidden = (const float*)d_in[0];
  // d_in[1] attention_mask (analytic causal, unused), d_in[2] position_ids (arange, unused)
  const float* Wq = (const float*)d_in[3];
  const float* Wk = (const float*)d_in[4];
  const float* Wv = (const float*)d_in[5];
  const float* Wo = (const float*)d_in[6];
  float* out = (float*)d_out;

  char* ws = (char*)d_ws;
  size_t off = 0;
  auto alloc = [&](size_t bytes) -> void* {
    void* p = ws + off;
    off += (bytes + 255) & ~(size_t)255;
    return p;
  };
  unsigned short* hbf    = (unsigned short*)alloc((size_t)TOKENS * HID * 2);
  unsigned short* WqkvT  = (unsigned short*)alloc((size_t)QKVN * HID * 2);
  unsigned short* WoT    = (unsigned short*)alloc((size_t)HID * HID * 2);
  unsigned short* qkv    = (unsigned short*)alloc((size_t)TOKENS * QKVN * 2);
  unsigned short* Qbuf   = (unsigned short*)alloc((size_t)BATCH * NHEADS * SEQ * HD * 2);
  unsigned short* Kbuf   = (unsigned short*)alloc((size_t)BATCH * NKV * SEQ * HD * 2);
  unsigned short* VtBuf  = (unsigned short*)alloc((size_t)BATCH * NKV * HD * SEQ * 2);
  unsigned short* attnO  = (unsigned short*)alloc((size_t)TOKENS * NHEADS * HD * 2);

  // 1) hidden f32 -> bf16
  {
    int n = TOKENS * HID;
    k_f32_to_bf16<<<(n + 255) / 256, 256, 0, stream>>>(hidden, hbf, n);
  }
  // 2) transpose weights into [N][K] bf16 (Wq/Wk/Wv stacked; Wo separate)
  {
    int n = HID * (NHEADS * HD);
    k_transpose_bf16<<<(n + 255) / 256, 256, 0, stream>>>(Wq, WqkvT, HID, NHEADS * HD);
    n = HID * (NKV * HD);
    k_transpose_bf16<<<(n + 255) / 256, 256, 0, stream>>>(
        Wk, WqkvT + (size_t)(NHEADS * HD) * HID, HID, NKV * HD);
    k_transpose_bf16<<<(n + 255) / 256, 256, 0, stream>>>(
        Wv, WqkvT + (size_t)(NHEADS * HD + NKV * HD) * HID, HID, NKV * HD);
    n = HID * HID;
    k_transpose_bf16<<<(n + 255) / 256, 256, 0, stream>>>(Wo, WoT, HID, HID);
  }
  // 3) fused QKV projection GEMM: [TOKENS, HID] x [HID, QKVN] -> bf16
  {
    dim3 grid(QKVN / 128, TOKENS / 128);
    k_gemm_bf16<true><<<grid, 256, 0, stream>>>(hbf, WqkvT, qkv, TOKENS, QKVN, HID);
  }
  // 4) RoPE on Q and K; transpose V
  {
    int nq = BATCH * SEQ * NHEADS * 64;
    k_rope<<<(nq + 255) / 256, 256, 0, stream>>>(qkv, Qbuf, NHEADS, 0, nq);
    int nk = BATCH * SEQ * NKV * 64;
    k_rope<<<(nk + 255) / 256, 256, 0, stream>>>(qkv, Kbuf, NKV, NHEADS * HD, nk);
    int nv = BATCH * SEQ * NKV * HD;
    k_scatter_v<<<(nv + 255) / 256, 256, 0, stream>>>(qkv, VtBuf, nv);
  }
  // 5) flash attention (soft-cap + causal + GQA)
  {
    int nblocks = BATCH * NHEADS * (SEQ / 16);  // 8192 waves
    k_attn<<<nblocks, 32, 0, stream>>>(Qbuf, Kbuf, VtBuf, attnO);
  }
  // 6) output projection GEMM -> f32 d_out
  {
    dim3 grid(HID / 128, TOKENS / 128);
    k_gemm_bf16<false><<<grid, 256, 0, stream>>>(attnO, WoT, out, TOKENS, HID, HID);
  }
}
